// MultiHeadAttention_56650618634930
// MI455X (gfx1250) — compile-verified, matrix operand order checked by reading, not run on hardware
//
#include <hip/hip_runtime.h>

// MI455X / gfx1250 wave32 WMMA multi-head attention.
// - All GEMMs: v_wmma_f32_16x16x32_f16 (f32 accumulate)
// - Global->LDS tile fills: GLOBAL_LOAD_ASYNC_TO_LDS_B128 (ASYNCcnt),
//   double-buffered so copy latency overlaps the WMMA phase.
// - Flash-attention online softmax (no SxS materialization).

typedef __attribute__((ext_vector_type(16))) _Float16 v16h;
typedef __attribute__((ext_vector_type(8)))  _Float16 v8h;
typedef __attribute__((ext_vector_type(8)))  float    v8f;

#define NB   4
#define NS   2048
#define ND   1024
#define NH   16
#define NHD  64
#define TILE_ELEMS (64 * 72)                    // one LDS tile buffer
#define LNROPE_OVER_HALF 0.28782313662425572f   // ln(10000)/32

// ---------------------------------------------------------------------------
// CDNA5 async global->LDS copy (GLOBAL_LOAD_ASYNC_TO_LDS_B128, ASYNCcnt).
// Per-lane: 16 bytes from the lane's global address to the lane's LDS address.
// Generic pointers to __shared__ carry the addrspace(3) byte offset in their
// low 32 bits (ISA 10.2), which is exactly what VDST wants.
// ---------------------------------------------------------------------------
__device__ __forceinline__ void async_copy16(void* lds_dst, const void* gsrc) {
  unsigned l = (unsigned)(uintptr_t)lds_dst;
  asm volatile("global_load_async_to_lds_b128 %0, %1, off"
               :
               : "v"(l), "v"(gsrc)
               : "memory");
}
__device__ __forceinline__ void async_wait0() {
  asm volatile("s_wait_asynccnt 0x0" ::: "memory");
}

// ---------------------------------------------------------------------------
// Fragment loader: tile stored row-major over contraction dim K, stride 72 f16.
// A-matrix 16x32 f16 layout (ISA 7.12.2): lane half=l>>4, idx=l&15:
//   VGPR0-3 : K = half*8 .. half*8+7   (packed pairs)
//   VGPR4-7 : K = 16+half*8 .. 16+half*8+7
// B-matrix 32x16 is the mirror (N across lanes) -> identical addressing when
// the B tile is stored as [n][k]. Two 16-byte ds loads per fragment.
// ---------------------------------------------------------------------------
__device__ __forceinline__ v16h load_frag(const _Float16* __restrict__ tile,
                                          int row_base, int k_base, int stride) {
  const int lane = threadIdx.x & 31;
  const int half = lane >> 4;
  const int idx  = lane & 15;
  const _Float16* p = tile + (row_base + idx) * stride + k_base + half * 8;
  v8h a = *(const v8h*)p;
  v8h b = *(const v8h*)(p + 16);
  v16h r;
#pragma unroll
  for (int i = 0; i < 8; ++i) { r[i] = a[i]; r[i + 8] = b[i]; }
  return r;
}

__device__ __forceinline__ v8f wmma_f16(v16h a, v16h b, v8f c) {
  return __builtin_amdgcn_wmma_f32_16x16x32_f16(false, a, false, b,
                                                (short)0, c, false, false);
}

// ---------------------------------------------------------------------------
// fp32 -> f16 conversion
// ---------------------------------------------------------------------------
__global__ void cvt_f32_to_f16_kernel(const float* __restrict__ src,
                                      _Float16* __restrict__ dst, int n) {
  int i = blockIdx.x * blockDim.x + threadIdx.x;
  if (i < n) dst[i] = (_Float16)src[i];
}

// ---------------------------------------------------------------------------
// Fused QKV projection + bias + RoPE.
// grid = (ND/64, (NB*NS)/64, 3)  z: 0=Q 1=K 2=V. block = 256 (8 waves).
// 64 tokens x 64 out-features per block (one head's dims, hd==64).
// Double-buffered 64x64 K-tiles filled by async-to-LDS copies.
// Q,K out: [B,H,S,hd] f16 (RoPE applied).  V out: [B,H,hd,S] f16 (transposed).
// ---------------------------------------------------------------------------
__global__ __launch_bounds__(256) void qkv_rope_kernel(
    const _Float16* __restrict__ X,                       // [NB*NS][ND]
    const _Float16* __restrict__ Wq, const _Float16* __restrict__ Wk,
    const _Float16* __restrict__ Wv,
    const float* __restrict__ bq, const float* __restrict__ bk,
    const float* __restrict__ bv,
    _Float16* __restrict__ Qo, _Float16* __restrict__ Ko,
    _Float16* __restrict__ Vt) {
  __shared__ __align__(16) _Float16 At[2 * TILE_ELEMS];
  __shared__ __align__(16) _Float16 Bt[2 * TILE_ELEMS];
  __shared__ __align__(16) float    Ct[64 * 68];

  const int mat = blockIdx.z;
  const _Float16* W  = (mat == 0) ? Wq : (mat == 1) ? Wk : Wv;
  const float* bias  = (mat == 0) ? bq : (mat == 1) ? bk : bv;
  const int m0 = blockIdx.y * 64;          // token base
  const int n0 = blockIdx.x * 64;          // out-feature base (== head*64)
  const int t  = threadIdx.x;
  const int wave = t >> 5, lane = t & 31;
  const int r0 = 16 * (wave & 3), c0 = 32 * (wave >> 2);
  const int half = lane >> 4, idx = lane & 15;
  const int lrow = t >> 2, lchunk = (t & 3) * 16;  // 64 rows x 4 chunks of 16
  const int ldsoff = lrow * 72 + lchunk;

  const _Float16* Xrow = X + (size_t)(m0 + lrow) * ND + lchunk;
  const _Float16* Wrow = W + (size_t)(n0 + lrow) * ND + lchunk;

  // Prologue: stream first 64-K slab into buffer 0.
  async_copy16(At + ldsoff,     Xrow);
  async_copy16(At + ldsoff + 8, Xrow + 8);
  async_copy16(Bt + ldsoff,     Wrow);
  async_copy16(Bt + ldsoff + 8, Wrow + 8);

  v8f acc0 = {}; v8f acc1 = {};
  for (int k0 = 0; k0 < ND; k0 += 64) {
    const int cur = (k0 >> 6) & 1;
    const _Float16* Ab = At + cur * TILE_ELEMS;
    const _Float16* Bb = Bt + cur * TILE_ELEMS;
    async_wait0();        // our copies for 'cur' have landed
    __syncthreads();      // everyone's landed; prior reads of 'next' retired
    if (k0 + 64 < ND) {   // overlap next slab's copy with this slab's WMMAs
      _Float16* An = At + (cur ^ 1) * TILE_ELEMS;
      _Float16* Bn = Bt + (cur ^ 1) * TILE_ELEMS;
      async_copy16(An + ldsoff,     Xrow + k0 + 64);
      async_copy16(An + ldsoff + 8, Xrow + k0 + 64 + 8);
      async_copy16(Bn + ldsoff,     Wrow + k0 + 64);
      async_copy16(Bn + ldsoff + 8, Wrow + k0 + 64 + 8);
    }
#pragma unroll
    for (int kk = 0; kk < 64; kk += 32) {
      v16h a  = load_frag(Ab, r0, kk, 72);
      v16h b0 = load_frag(Bb, c0, kk, 72);
      v16h b1 = load_frag(Bb, c0 + 16, kk, 72);
      acc0 = wmma_f16(a, b0, acc0);
      acc1 = wmma_f16(a, b1, acc1);
    }
  }

  // Stage (with bias) into LDS so the rotate-half pairing is easy.
#pragma unroll
  for (int v = 0; v < 8; ++v) {
    int row = r0 + v + 8 * half;
    Ct[row * 68 + c0 + idx]      = acc0[v] + bias[n0 + c0 + idx];
    Ct[row * 68 + c0 + 16 + idx] = acc1[v] + bias[n0 + c0 + 16 + idx];
  }
  __syncthreads();

  const int h = n0 >> 6;                   // head index
  const int row = t >> 2;
  const int token = m0 + row;
  const int bb  = token >> 11;             // /2048
  const int pos = token & 2047;
#pragma unroll
  for (int e = 0; e < 16; ++e) {
    int j = (t & 3) * 16 + e;              // dim within head
    float val = Ct[row * 68 + j];
    if (mat < 2) {
      int   pj  = (j < 32) ? j + 32 : j - 32;
      float pv  = Ct[row * 68 + pj];
      float rot = (j < 32) ? -pv : pv;
      float fr  = (float)pos * expf(-(float)(j & 31) * LNROPE_OVER_HALF);
      val = val * cosf(fr) + rot * sinf(fr);
      _Float16 hv = (_Float16)val;
      size_t o = (((size_t)bb * NH + h) * NS + pos) * NHD + j;
      if (mat == 0) Qo[o] = hv; else Ko[o] = hv;
    } else {
      Vt[(((size_t)bb * NH + h) * NHD + j) * NS + pos] = (_Float16)val;
    }
  }
}

// ---------------------------------------------------------------------------
// Flash attention. grid = (NS/64, NB*NH), block = 256.
// 64-query tile per block, sweeping 64-key blocks with online softmax.
// K/V tiles double-buffered via async-to-LDS; scores and PV via WMMA.
// ---------------------------------------------------------------------------
__global__ __launch_bounds__(256) void flash_attn_kernel(
    const _Float16* __restrict__ Q, const _Float16* __restrict__ K,
    const _Float16* __restrict__ Vt, const float* __restrict__ mask,
    _Float16* __restrict__ ctx) {
  __shared__ __align__(16) _Float16 Qt[TILE_ELEMS];
  __shared__ __align__(16) _Float16 Kt[2 * TILE_ELEMS];
  __shared__ __align__(16) _Float16 Vv[2 * TILE_ELEMS];   // [d][k_local]
  __shared__ __align__(16) _Float16 Pt[TILE_ELEMS];
  __shared__ __align__(16) float    St[64 * 68];
  __shared__ float m_s[64], l_s[64], sc_s[64];

  const int bh = blockIdx.y;
  const int bb = bh >> 4;                    // batch
  const int h  = bh & 15;                    // head
  const int q0 = blockIdx.x * 64;
  const int t  = threadIdx.x;
  const int wave = t >> 5, lane = t & 31;
  const int r0 = 16 * (wave & 3), c0 = 32 * (wave >> 2);
  const int half = lane >> 4, idx = lane & 15;
  const int lrow = t >> 2, lchunk = (t & 3) * 16;  // 64 rows x 4 chunks of 16
  const int ldsoff = lrow * 72 + lchunk;

  const _Float16* Kbase = K  + ((size_t)bh * NS  + lrow) * NHD + lchunk;
  const _Float16* Vbase = Vt + ((size_t)bh * NHD + lrow) * NS  + lchunk;

  // Prologue: key/value block 0 into buffer 0.
  async_copy16(Kt + ldsoff,     Kbase);
  async_copy16(Kt + ldsoff + 8, Kbase + 8);
  async_copy16(Vv + ldsoff,     Vbase);
  async_copy16(Vv + ldsoff + 8, Vbase + 8);

  {
    const _Float16* s = Q + ((size_t)bh * NS + q0 + lrow) * NHD + lchunk;
    *(v8h*)(Qt + ldsoff)     = *(const v8h*)s;
    *(v8h*)(Qt + ldsoff + 8) = *(const v8h*)(s + 8);
  }
  if (t < 64) { m_s[t] = -1e30f; l_s[t] = 0.0f; }
  v8f o0 = {}; v8f o1 = {};

  for (int kb = 0; kb < NS / 64; ++kb) {
    const int cur = kb & 1;
    const _Float16* Kb = Kt + cur * TILE_ELEMS;
    const _Float16* Vb = Vv + cur * TILE_ELEMS;
    async_wait0();
    __syncthreads();
    if (kb + 1 < NS / 64) {   // stream next K/V block behind this one's math
      _Float16* kn = Kt + (cur ^ 1) * TILE_ELEMS;
      _Float16* vn = Vv + (cur ^ 1) * TILE_ELEMS;
      const _Float16* ksrc = Kbase + (size_t)(kb + 1) * 64 * NHD;
      const _Float16* vsrc = Vbase + (kb + 1) * 64;
      async_copy16(kn + ldsoff,     ksrc);
      async_copy16(kn + ldsoff + 8, ksrc + 8);
      async_copy16(vn + ldsoff,     vsrc);
      async_copy16(vn + ldsoff + 8, vsrc + 8);
    }

    // S = Q . K^T  (contraction over d = 64)
    v8f s0 = {}; v8f s1 = {};
#pragma unroll
    for (int kk = 0; kk < 64; kk += 32) {
      v16h a  = load_frag(Qt, r0, kk, 72);
      v16h b0 = load_frag(Kb, c0, kk, 72);
      v16h b1 = load_frag(Kb, c0 + 16, kk, 72);
      s0 = wmma_f16(a, b0, s0);
      s1 = wmma_f16(a, b1, s1);
    }
#pragma unroll
    for (int v = 0; v < 8; ++v) {
      int row = r0 + v + 8 * half;
      St[row * 68 + c0 + idx]      = s0[v];
      St[row * 68 + c0 + 16 + idx] = s1[v];
    }
    __syncthreads();

    // Online softmax update (one thread per query row).
    if (t < 64) {
      int r = t;
      const float* mrow = mask + ((size_t)bb * NS + q0 + r) * NS + kb * 64;
      float mx = m_s[r];
      for (int c = 0; c < 64; ++c) {
        float lg = St[r * 68 + c] * 0.125f + mrow[c];   // 1/sqrt(64)
        St[r * 68 + c] = lg;
        mx = fmaxf(mx, lg);
      }
      float sc = __expf(m_s[r] - mx);
      float ls = l_s[r] * sc;
      for (int c = 0; c < 64; ++c) {
        float p = __expf(St[r * 68 + c] - mx);
        ls += p;
        Pt[r * 72 + c] = (_Float16)p;
      }
      m_s[r] = mx; l_s[r] = ls; sc_s[r] = sc;
    }
    __syncthreads();

    // Rescale accumulators, then O += P . V  (contraction over k_local = 64)
#pragma unroll
    for (int v = 0; v < 8; ++v) {
      float sc = sc_s[r0 + v + 8 * half];
      o0[v] *= sc; o1[v] *= sc;
    }
#pragma unroll
    for (int kk = 0; kk < 64; kk += 32) {
      v16h a  = load_frag(Pt, r0, kk, 72);
      v16h b0 = load_frag(Vb, c0, kk, 72);
      v16h b1 = load_frag(Vb, c0 + 16, kk, 72);
      o0 = wmma_f16(a, b0, o0);
      o1 = wmma_f16(a, b1, o1);
    }
    // no trailing barrier: next iteration's wait+barrier protects buffers
  }

  // Normalize and store ctx as [b][s][h][hd] f16.
#pragma unroll
  for (int v = 0; v < 8; ++v) {
    int row = r0 + v + 8 * half;
    float inv = 1.0f / l_s[row];
    size_t base = (((size_t)bb * NS + q0 + row) * NH + h) * NHD;
    ctx[base + c0 + idx]      = (_Float16)(o0[v] * inv);
    ctx[base + c0 + 16 + idx] = (_Float16)(o1[v] * inv);
  }
}

// ---------------------------------------------------------------------------
// Output projection: out(f32) = ctx(f16)[M][K] . Wo(f16)[N][K]^T + bo.
// grid = (ND/64, (NB*NS)/64), block = 256. Double-buffered async tiles.
// ---------------------------------------------------------------------------
__global__ __launch_bounds__(256) void out_proj_kernel(
    const _Float16* __restrict__ A, const _Float16* __restrict__ W,
    const float* __restrict__ bias, float* __restrict__ out) {
  __shared__ __align__(16) _Float16 At[2 * TILE_ELEMS];
  __shared__ __align__(16) _Float16 Bt[2 * TILE_ELEMS];
  const int m0 = blockIdx.y * 64;
  const int n0 = blockIdx.x * 64;
  const int t  = threadIdx.x;
  const int wave = t >> 5, lane = t & 31;
  const int r0 = 16 * (wave & 3), c0 = 32 * (wave >> 2);
  const int half = lane >> 4, idx = lane & 15;
  const int lrow = t >> 2, lchunk = (t & 3) * 16;
  const int ldsoff = lrow * 72 + lchunk;

  const _Float16* Arow = A + (size_t)(m0 + lrow) * ND + lchunk;
  const _Float16* Wrow = W + (size_t)(n0 + lrow) * ND + lchunk;

  async_copy16(At + ldsoff,     Arow);
  async_copy16(At + ldsoff + 8, Arow + 8);
  async_copy16(Bt + ldsoff,     Wrow);
  async_copy16(Bt + ldsoff + 8, Wrow + 8);

  v8f acc0 = {}; v8f acc1 = {};
  for (int k0 = 0; k0 < ND; k0 += 64) {
    const int cur = (k0 >> 6) & 1;
    const _Float16* Ab = At + cur * TILE_ELEMS;
    const _Float16* Bb = Bt + cur * TILE_ELEMS;
    async_wait0();
    __syncthreads();
    if (k0 + 64 < ND) {
      _Float16* An = At + (cur ^ 1) * TILE_ELEMS;
      _Float16* Bn = Bt + (cur ^ 1) * TILE_ELEMS;
      async_copy16(An + ldsoff,     Arow + k0 + 64);
      async_copy16(An + ldsoff + 8, Arow + k0 + 64 + 8);
      async_copy16(Bn + ldsoff,     Wrow + k0 + 64);
      async_copy16(Bn + ldsoff + 8, Wrow + k0 + 64 + 8);
    }
#pragma unroll
    for (int kk = 0; kk < 64; kk += 32) {
      v16h a  = load_frag(Ab, r0, kk, 72);
      v16h b0 = load_frag(Bb, c0, kk, 72);
      v16h b1 = load_frag(Bb, c0 + 16, kk, 72);
      acc0 = wmma_f16(a, b0, acc0);
      acc1 = wmma_f16(a, b1, acc1);
    }
  }
#pragma unroll
  for (int v = 0; v < 8; ++v) {
    int row = m0 + r0 + v + 8 * half;
    out[(size_t)row * ND + n0 + c0 + idx] =
        acc0[v] + bias[n0 + c0 + idx];
    out[(size_t)row * ND + n0 + c0 + 16 + idx] =
        acc1[v] + bias[n0 + c0 + 16 + idx];
  }
}

// ---------------------------------------------------------------------------
// Host launcher
// ---------------------------------------------------------------------------
extern "C" void kernel_launch(void* const* d_in, const int* in_sizes, int n_in,
                              void* d_out, int out_size, void* d_ws, size_t ws_size,
                              hipStream_t stream) {
  (void)in_sizes; (void)n_in; (void)out_size; (void)ws_size;
  const float* hidden = (const float*)d_in[0];
  const float* mask   = (const float*)d_in[1];
  const float* Wq = (const float*)d_in[2];
  const float* bq = (const float*)d_in[3];
  const float* Wk = (const float*)d_in[4];
  const float* bk = (const float*)d_in[5];
  const float* Wv = (const float*)d_in[6];
  const float* bv = (const float*)d_in[7];
  const float* Wo = (const float*)d_in[8];
  const float* bo = (const float*)d_in[9];
  float* out = (float*)d_out;

  const size_t nTok   = (size_t)NB * NS;          // 8192
  const size_t nHid   = nTok * ND;                // 8388608
  const size_t nW     = (size_t)ND * ND;          // 1048576

  char* ws = (char*)d_ws;
  size_t off = 0;
  _Float16* hidden_h = (_Float16*)(ws + off); off += nHid * 2;
  _Float16* Wq_h     = (_Float16*)(ws + off); off += nW * 2;
  _Float16* Wk_h     = (_Float16*)(ws + off); off += nW * 2;
  _Float16* Wv_h     = (_Float16*)(ws + off); off += nW * 2;
  _Float16* Wo_h     = (_Float16*)(ws + off); off += nW * 2;
  _Float16* Qh       = (_Float16*)(ws + off); off += nHid * 2;
  _Float16* Kh       = (_Float16*)(ws + off); off += nHid * 2;
  _Float16* Vth      = (_Float16*)(ws + off); off += nHid * 2;
  _Float16* ctx      = (_Float16*)(ws + off); off += nHid * 2;

  const int cvtB = 256;
  cvt_f32_to_f16_kernel<<<(unsigned)((nHid + cvtB - 1) / cvtB), cvtB, 0, stream>>>(hidden, hidden_h, (int)nHid);
  cvt_f32_to_f16_kernel<<<(unsigned)((nW + cvtB - 1) / cvtB), cvtB, 0, stream>>>(Wq, Wq_h, (int)nW);
  cvt_f32_to_f16_kernel<<<(unsigned)((nW + cvtB - 1) / cvtB), cvtB, 0, stream>>>(Wk, Wk_h, (int)nW);
  cvt_f32_to_f16_kernel<<<(unsigned)((nW + cvtB - 1) / cvtB), cvtB, 0, stream>>>(Wv, Wv_h, (int)nW);
  cvt_f32_to_f16_kernel<<<(unsigned)((nW + cvtB - 1) / cvtB), cvtB, 0, stream>>>(Wo, Wo_h, (int)nW);

  dim3 gQKV(ND / 64, (unsigned)(nTok / 64), 3);
  qkv_rope_kernel<<<gQKV, 256, 0, stream>>>(hidden_h, Wq_h, Wk_h, Wv_h,
                                            bq, bk, bv, Qh, Kh, Vth);

  dim3 gAtt(NS / 64, NB * NH);
  flash_attn_kernel<<<gAtt, 256, 0, stream>>>(Qh, Kh, Vth, mask, ctx);

  dim3 gProj(ND / 64, (unsigned)(nTok / 64));
  out_proj_kernel<<<gProj, 256, 0, stream>>>(ctx, Wo_h, bo, out);
}
